// Model_57131654972138
// MI455X (gfx1250) — matrix operand order repaired and checked by reading, not compiled, not run on hardware
//
#include <hip/hip_runtime.h>
#include <hip/hip_bf16.h>

// ---------------------------------------------------------------------------
// Problem constants (from reference)
// ---------------------------------------------------------------------------
#define NU   50000
#define NM   20000
#define FEAT 32
#define NE   262144
#define EL   100000
#define H    512
#define ED   7
#define NC   7

typedef __attribute__((ext_vector_type(16))) __bf16 v16bf;
typedef __attribute__((ext_vector_type(8)))  float  v8f;

// ---------------------------------------------------------------------------
// Transpose + cast weights once:  in f32 [K x N]  ->  outT bf16 [N x K]
// ---------------------------------------------------------------------------
__global__ __launch_bounds__(256) void transpose_cast_bf16(
    const float* __restrict__ in, __bf16* __restrict__ outT, int K, int N)
{
    __shared__ float t[32][33];
    const int k0 = blockIdx.y * 32, n0 = blockIdx.x * 32;
    const int tx = threadIdx.x & 31, ty = threadIdx.x >> 5;   // ty: 0..7
    #pragma unroll
    for (int i = ty; i < 32; i += 8)
        t[i][tx] = in[(size_t)(k0 + i) * N + n0 + tx];
    __syncthreads();
    #pragma unroll
    for (int i = ty; i < 32; i += 8)
        outT[(size_t)(n0 + i) * K + k0 + tx] = (__bf16)t[tx][i];
}

// ---------------------------------------------------------------------------
// Fused WMMA GEMM:
//   out = relu?( (A*rowscale) @ B  (+ A2 @ B2)  + bias + extra )
// A,A2: f32 row-major [M x K]; Bt,B2t: bf16 TRANSPOSED [N x K]; out f32 [M x N]
// Block: 256 threads (8 wave32s), tile 256(M) x 64(N), K stepped by 32.
// Each wave computes 32(M) x 64(N): two A-frags share four B-frags
// -> 8 v_wmma per 12 ds_load_b128.
// B slab staged via global_load_async_to_lds_b128 (ASYNCcnt), overlapping the
// VALU f32->bf16 conversion of the A slab.
// ---------------------------------------------------------------------------
#define BM 256
#define BN 64
#define SA 40   // LDS row stride (bf16) = 80B: 16B-aligned, spreads 64 banks
#define SB 40

template<bool RS>
__global__ __launch_bounds__(256) void wmma_gemm_fused(
    const float* __restrict__ A,  const __bf16* __restrict__ Bt,
    const float* __restrict__ rowscale,
    const float* __restrict__ A2, const __bf16* __restrict__ B2t,
    const float* __restrict__ bias, const float* __restrict__ extra,
    float* __restrict__ out, int M, int N, int K, int K2, int relu)
{
    __shared__ __align__(16) __bf16 sA[BM * SA];   // [row][k]  (20 KB)
    __shared__ __align__(16) __bf16 sBt[BN * SB];  // [col][k]  ( 5 KB)

    const int tid  = threadIdx.x;
    const int wv   = tid >> 5;
    const int lane = tid & 31;
    const int rowBase = blockIdx.y * BM;
    const int colBase = blockIdx.x * BN;

    v8f acc0[4] = {};   // rows [wv*32 .. wv*32+16)
    v8f acc1[4] = {};   // rows [wv*32+16 .. wv*32+32)

    for (int pass = 0; pass < 2; ++pass) {
        const float*  Ap  = pass ? A2  : A;
        const __bf16* Btp = pass ? B2t : Bt;
        const int     Kp  = pass ? K2  : K;
        if (!Ap) continue;

        for (int k0 = 0; k0 < Kp; k0 += 32) {
            __syncthreads();   // previous iteration done reading LDS

            // ---- stage B slab: async copy BT[colBase..+64)[k0..k0+32) ----
            {
                const int c    = tid >> 2;          // 0..63 (column in tile)
                const int part = tid & 3;           // 4 x 8 bf16 = one row
                const __bf16* gsrc = Btp + (size_t)(colBase + c) * Kp + k0 + part * 8;
                unsigned ldsa = (unsigned)(uintptr_t)&sBt[c * SB + part * 8];
                asm volatile("global_load_async_to_lds_b128 %0, %1, off"
                             :: "v"(ldsa), "v"(gsrc) : "memory");
            }

            // ---- stage A slab: 256x32 f32 -> bf16 (branchless row guard) ----
            const int kpf = (k0 + 32 < Kp) ? k0 + 32 : k0;  // branchless prefetch addr
            #pragma unroll
            for (int q = 0; q < 8; ++q) {
                int idx  = tid + 256 * q;           // 0..2047
                int r    = idx >> 3;                // 0..255
                int jc   = (idx & 7) * 4;           // 0,4,..,28
                int grow = rowBase + r;
                int gc   = grow < M ? grow : M - 1; // clamp: load always legal
                float msk = grow < M ? 1.0f : 0.0f;
                float4 v = *(const float4*)(Ap + (size_t)gc * Kp + k0 + jc);
                float s  = msk;
                if (RS && pass == 0) s *= rowscale[gc];
                v.x *= s; v.y *= s; v.z *= s; v.w *= s;
                __bf16 tmp[4] = {(__bf16)v.x, (__bf16)v.y, (__bf16)v.z, (__bf16)v.w};
                *(uint2*)&sA[r * SA + jc] = *(const uint2*)tmp;
                __builtin_prefetch(Ap + (size_t)gc * Kp + kpf + jc, 0, 1);
            }

            asm volatile("s_wait_asynccnt 0x0" ::: "memory");
            __syncthreads();

            // ---- A fragments (ISA 7.12.2, 16-bit A 16x32) ----
            // lane<16: row=lane, K={0..7,16..23}; lane>=16: row=lane-16, K={8..15,24..31}
            const int r   = lane & 15;
            const int kbA = (lane & 16) ? 8 : 0;
            const __bf16* ap0 = &sA[(wv * 32 + r) * SA + kbA];
            const __bf16* ap1 = ap0 + 16 * SA;
            v16bf af0, af1;
            ((uint4*)&af0)[0] = *(const uint4*)(ap0);
            ((uint4*)&af0)[1] = *(const uint4*)(ap0 + 16);
            ((uint4*)&af1)[0] = *(const uint4*)(ap1);
            ((uint4*)&af1)[1] = *(const uint4*)(ap1 + 16);

            // ---- B fragments: lane<16: col=lane, K={0..7,16..23}; lane>=16: K={8..15,24..31} ----
            const int kbB = (lane & 16) ? 8 : 0;
            #pragma unroll
            for (int t = 0; t < 4; ++t) {
                const __bf16* bp = &sBt[(t * 16 + r) * SB + kbB];
                v16bf bfrag;
                ((uint4*)&bfrag)[0] = *(const uint4*)(bp);
                ((uint4*)&bfrag)[1] = *(const uint4*)(bp + 16);
                acc0[t] = __builtin_amdgcn_wmma_f32_16x16x32_bf16(
                    false, af0, false, bfrag, (short)0, acc0[t], false, false);
                acc1[t] = __builtin_amdgcn_wmma_f32_16x16x32_bf16(
                    false, af1, false, bfrag, (short)0, acc1[t], false, false);
            }
        }
    }

    // ---- epilogue: C/D layout: VGPR e -> M=e (+8 hi half), N = lane&15 ----
    const int r    = lane & 15;
    const int rOff = (lane & 16) ? 8 : 0;
    #pragma unroll
    for (int t = 0; t < 4; ++t) {
        int col  = colBase + t * 16 + r;
        float bv = bias ? bias[col] : 0.0f;
        #pragma unroll
        for (int e = 0; e < 8; ++e) {
            int grow0 = rowBase + wv * 32 + e + rOff;
            int grow1 = grow0 + 16;
            if (grow0 < M) {
                float v = acc0[t][e] + bv;
                if (extra) v += extra[(size_t)grow0 * N + col];
                if (relu)  v = fmaxf(v, 0.0f);
                out[(size_t)grow0 * N + col] = v;
            }
            if (grow1 < M) {
                float v = acc1[t][e] + bv;
                if (extra) v += extra[(size_t)grow1 * N + col];
                if (relu)  v = fmaxf(v, 0.0f);
                out[(size_t)grow1 * N + col] = v;
            }
        }
    }
}

// ---------------------------------------------------------------------------
// Per-edge message + scatter-add:  agg[dst] += x_src[src] + ea@ew + eb
// One wave32 per edge; per-lane float4 over the 512 channels.
// ---------------------------------------------------------------------------
__global__ __launch_bounds__(256) void edge_msg_scatter(
    const float* __restrict__ xsrc, const int* __restrict__ ei,
    const float* __restrict__ ea,   const float* __restrict__ ew,
    const float* __restrict__ eb,   float* __restrict__ agg, int E)
{
    const int wv   = threadIdx.x >> 5;
    const int lane = threadIdx.x & 31;
    const int e    = blockIdx.x * 8 + wv;
    if (e >= E) return;
    const int s = ei[e];
    const int d = ei[E + e];
    const float* av = ea + (size_t)e * ED;
    float a[ED];
    #pragma unroll
    for (int q = 0; q < ED; ++q) a[q] = av[q];
    const float* xs = xsrc + (size_t)s * H;
    float*       ag = agg  + (size_t)d * H;
    #pragma unroll
    for (int j = 0; j < 4; ++j) {
        int c = 4 * lane + 128 * j;
        float4 x = *(const float4*)(xs + c);
        float4 b = *(const float4*)(eb + c);
        float4 m = make_float4(x.x + b.x, x.y + b.y, x.z + b.z, x.w + b.w);
        #pragma unroll
        for (int q = 0; q < ED; ++q) {
            float4 w = *(const float4*)(ew + (size_t)q * H + c);
            m.x += a[q] * w.x; m.y += a[q] * w.y;
            m.z += a[q] * w.z; m.w += a[q] * w.w;
        }
        atomicAdd(&ag[c + 0], m.x);
        atomicAdd(&ag[c + 1], m.y);
        atomicAdd(&ag[c + 2], m.z);
        atomicAdd(&ag[c + 3], m.w);
    }
}

__global__ void count_dst(const int* __restrict__ dst, float* __restrict__ cnt, int E) {
    int i = blockIdx.x * 256 + threadIdx.x;
    if (i < E) atomicAdd(&cnt[dst[i]], 1.0f);
}

__global__ void recip_clamp(float* __restrict__ c, int n) {
    int i = blockIdx.x * 256 + threadIdx.x;
    if (i < n) c[i] = 1.0f / fmaxf(c[i], 1.0f);
}

// ---------------------------------------------------------------------------
// Edge classifier: out[l] = [u2[ui], m2[mi]] @ cls_w + cls_b   (C=7)
// One wave32 per labeled edge; shuffle reduction.
// ---------------------------------------------------------------------------
__global__ __launch_bounds__(256) void classifier(
    const float* __restrict__ u2, const float* __restrict__ m2,
    const float* __restrict__ clsw, const float* __restrict__ clsb,
    const int* __restrict__ eli, float* __restrict__ out, int L)
{
    const int wv   = threadIdx.x >> 5;
    const int lane = threadIdx.x & 31;
    const int l    = blockIdx.x * 8 + wv;
    if (l >= L) return;
    const float* ur = u2 + (size_t)eli[l]     * H;
    const float* mr = m2 + (size_t)eli[L + l] * H;
    float acc[NC] = {};
    #pragma unroll
    for (int j = 0; j < 4; ++j) {
        int c = 4 * lane + 128 * j;
        float4 uv = *(const float4*)(ur + c);
        float4 mv = *(const float4*)(mr + c);
        #pragma unroll
        for (int i = 0; i < 4; ++i) {
            float u = (&uv.x)[i], m = (&mv.x)[i];
            const float* w0 = clsw + (size_t)(c + i) * NC;
            const float* w1 = clsw + (size_t)(H + c + i) * NC;
            #pragma unroll
            for (int q = 0; q < NC; ++q) acc[q] += u * w0[q] + m * w1[q];
        }
    }
    #pragma unroll
    for (int off = 16; off; off >>= 1)
        #pragma unroll
        for (int q = 0; q < NC; ++q) acc[q] += __shfl_down(acc[q], off, 32);
    if (lane == 0)
        #pragma unroll
        for (int q = 0; q < NC; ++q) out[(size_t)l * NC + q] = acc[q] + clsb[q];
}

// ---------------------------------------------------------------------------
extern "C" void kernel_launch(void* const* d_in, const int* in_sizes, int n_in,
                              void* d_out, int out_size, void* d_ws, size_t ws_size,
                              hipStream_t stream)
{
    (void)in_sizes; (void)n_in; (void)out_size; (void)ws_size;
    const float* movie_x     = (const float*)d_in[0];
    const float* user_emb    = (const float*)d_in[1];
    const float* movie_emb   = (const float*)d_in[2];
    const float* movie_lin_w = (const float*)d_in[3];
    const float* movie_lin_b = (const float*)d_in[4];
    const float* ea_um       = (const float*)d_in[5];
    const float* ea_mu       = (const float*)d_in[6];
    const float* lw[4], *lb[4], *rw[4], *ew[4], *ebv[4];
    for (int p = 0; p < 4; ++p) {
        lw[p]  = (const float*)d_in[7 + 5 * p + 0];
        lb[p]  = (const float*)d_in[7 + 5 * p + 1];
        rw[p]  = (const float*)d_in[7 + 5 * p + 2];
        ew[p]  = (const float*)d_in[7 + 5 * p + 3];
        ebv[p] = (const float*)d_in[7 + 5 * p + 4];
    }
    const float* cls_w = (const float*)d_in[27];
    const float* cls_b = (const float*)d_in[28];
    const int*   ei_um = (const int*)d_in[29];
    const int*   ei_mu = (const int*)d_in[30];
    const int*   eli   = (const int*)d_in[31];
    float* out = (float*)d_out;

    // ---- workspace carve-up ----
    char* p = (char*)d_ws;
    auto alloc = [&](size_t bytes) -> void* {
        void* r = (void*)p; p += (bytes + 255) & ~(size_t)255; return r;
    };
    __bf16* mlwT = (__bf16*)alloc((size_t)FEAT * H * 2);   // [H x FEAT]
    __bf16* lwT[4], *rwT[4];                                // [H x H] each
    for (int q = 0; q < 4; ++q) {
        lwT[q] = (__bf16*)alloc((size_t)H * H * 2);
        rwT[q] = (__bf16*)alloc((size_t)H * H * 2);
    }
    float* x_m   = (float*)alloc((size_t)NM * H * 4);
    float* u1    = (float*)alloc((size_t)NU * H * 4);
    float* m1    = (float*)alloc((size_t)NM * H * 4);
    float* u2    = (float*)alloc((size_t)NU * H * 4);
    float* m2    = (float*)alloc((size_t)NM * H * 4);
    float* agg_u = (float*)alloc((size_t)NU * H * 4);
    float* agg_m = (float*)alloc((size_t)NM * H * 4);
    float* inv_u = (float*)alloc((size_t)NU * 4);
    float* inv_m = (float*)alloc((size_t)NM * 4);

    // ---- weight transpose+cast to bf16 [N x K] ----
    transpose_cast_bf16<<<dim3(H / 32, FEAT / 32), 256, 0, stream>>>(
        movie_lin_w, mlwT, FEAT, H);
    for (int q = 0; q < 4; ++q) {
        transpose_cast_bf16<<<dim3(H / 32, H / 32), 256, 0, stream>>>(lw[q], lwT[q], H, H);
        transpose_cast_bf16<<<dim3(H / 32, H / 32), 256, 0, stream>>>(rw[q], rwT[q], H, H);
    }

    // ---- degree reciprocals (same edge lists both layers) ----
    hipMemsetAsync(inv_u, 0, (size_t)NU * 4, stream);
    hipMemsetAsync(inv_m, 0, (size_t)NM * 4, stream);
    count_dst<<<(NE + 255) / 256, 256, 0, stream>>>(ei_mu + NE, inv_u, NE);
    count_dst<<<(NE + 255) / 256, 256, 0, stream>>>(ei_um + NE, inv_m, NE);
    recip_clamp<<<(NU + 255) / 256, 256, 0, stream>>>(inv_u, NU);
    recip_clamp<<<(NM + 255) / 256, 256, 0, stream>>>(inv_m, NM);

    // ---- movie encoder: x_m = movie_x @ mlw + b + movie_emb ----
    {
        dim3 g(H / BN, (NM + BM - 1) / BM);
        wmma_gemm_fused<false><<<g, 256, 0, stream>>>(
            movie_x, mlwT, nullptr, nullptr, nullptr,
            movie_lin_b, movie_emb, x_m, NM, H, FEAT, 0, 0);
    }

    // conv: scatter then fused (agg*inv)@lw + x_dst@rw + lb
    auto conv = [&](const float* xsrc, const float* xdst, const int* ei,
                    const float* ea, int pIdx, const float* inv,
                    float* aggbuf, float* dst, int Mdst, int relu) {
        hipMemsetAsync(aggbuf, 0, (size_t)Mdst * H * 4, stream);
        edge_msg_scatter<<<(NE + 7) / 8, 256, 0, stream>>>(
            xsrc, ei, ea, ew[pIdx], ebv[pIdx], aggbuf, NE);
        dim3 g(H / BN, (Mdst + BM - 1) / BM);
        wmma_gemm_fused<true><<<g, 256, 0, stream>>>(
            aggbuf, lwT[pIdx], inv, xdst, rwT[pIdx],
            lb[pIdx], nullptr, dst, Mdst, H, H, H, relu);
    };

    // ---- layer 1 (ReLU) ----
    conv(x_m,      user_emb, ei_mu, ea_mu, 0, inv_u, agg_u, u1, NU, 1); // mu1
    conv(user_emb, x_m,      ei_um, ea_um, 1, inv_m, agg_m, m1, NM, 1); // um1
    // ---- layer 2 ----
    conv(m1, u1, ei_mu, ea_mu, 2, inv_u, agg_u, u2, NU, 0);             // mu2
    conv(u1, m1, ei_um, ea_um, 3, inv_m, agg_m, m2, NM, 0);             // um2

    // ---- classifier ----
    classifier<<<(EL + 7) / 8, 256, 0, stream>>>(u2, m2, cls_w, cls_b, eli, out, EL);
}
